// InterventionalAttention_2637109919798
// MI455X (gfx1250) — matrix operand order repaired and checked by reading, not compiled, not run on hardware
//
#include <hip/hip_runtime.h>

typedef _Float16 v16h __attribute__((ext_vector_type(16)));
typedef _Float16 v8h  __attribute__((ext_vector_type(8)));
typedef _Float16 v4h  __attribute__((ext_vector_type(4)));
typedef float    v8f  __attribute__((ext_vector_type(8)));
typedef float    f32x4 __attribute__((ext_vector_type(4)));
typedef unsigned int u32x4 __attribute__((ext_vector_type(4)));
typedef int          i32x4 __attribute__((ext_vector_type(4)));
typedef int          i32x8 __attribute__((ext_vector_type(8)));

#define DM    1024
#define HEADS 16
#define DK    64
#define SEQ   2048
#define BATCH 2
#define NTOK  (BATCH*SEQ)
#define LAM   0.5f

#if __has_builtin(__builtin_amdgcn_tensor_load_to_lds) && \
    __has_builtin(__builtin_amdgcn_s_wait_tensorcnt)
#define USE_TDM 1
#else
#define USE_TDM 0
#endif

__device__ __forceinline__ v8f wmma16(v16h a, v16h b, v8f c) {
  // D = A(16x32 f16) * B(32x16 f16) + C(16x16 f32)
  return __builtin_amdgcn_wmma_f32_16x16x32_f16(false, a, false, b, (short)0, c,
                                                false, false);
}

__device__ __forceinline__ v16h cat16(v8h lo, v8h hi) {
  v16h r;
#pragma unroll
  for (int i = 0; i < 8; ++i) { r[i] = lo[i]; r[i + 8] = hi[i]; }
  return r;
}

__device__ __forceinline__ v4h cvt4(f32x4 v) {
  v4h r;
#pragma unroll
  for (int i = 0; i < 4; ++i) r[i] = (_Float16)v[i];
  return r;
}

#if USE_TDM
// Issue a TDM load of a row-major f16 tile [64 rows][256 cols], row stride SEQ,
// from `gsrc` into LDS at byte offset `lds_off`.  D# layout per CDNA5 ISA ch8.
__device__ __forceinline__ void tdm_load_tile(const _Float16* gsrc,
                                              unsigned lds_off) {
  unsigned long long ga = (unsigned long long)(uintptr_t)gsrc;
  u32x4 g0;
  g0[0] = 1u;                                    // count=1, user mode
  g0[1] = lds_off;                               // lds_addr (bytes)
  g0[2] = (unsigned)(ga & 0xffffffffu);          // global_addr[31:0]
  g0[3] = (unsigned)((ga >> 32) & 0x01ffffffu)   // global_addr[56:32]
          | (2u << 30);                          // type = 2 ("image")
  i32x8 g1;
  g1[0] = 0x00010000;                            // data_size = 1 (2 bytes)
  g1[1] = (int)(2048u << 16);                    // tensor_dim0 = SEQ
  g1[2] = (int)(64u << 16);                      // tensor_dim1 = 64
  g1[3] = (int)(256u << 16);                     // tile_dim0 = 256 (contig)
  g1[4] = 64;                                    // tile_dim1 = 64 rows
  g1[5] = 2048;                                  // tensor_dim0_stride = SEQ
  g1[6] = 0;
  g1[7] = 0;
  i32x4 gz = {0, 0, 0, 0};
#if defined(__clang_major__) && __clang_major__ >= 23
  i32x8 gz8 = {0, 0, 0, 0, 0, 0, 0, 0};
  __builtin_amdgcn_tensor_load_to_lds(g0, g1, gz, gz, gz8, 0);
#else
  __builtin_amdgcn_tensor_load_to_lds(g0, g1, gz, gz, 0);
#endif
}
#endif

// ---------------------------------------------------------------------------
// Kernel 1: fused QKV projection. Y = x @ W^T + b, f16 output.
// Q,K stored [B,H,L,dk]; V stored transposed [B,H,dk,L] so the attention
// kernel can TDM-load V^T tiles as plain row-major 2D tiles.
// 128x64 tile per block, 8 waves, 4 WMMA tiles per wave per k-step.
// Global->register->LDS double buffering: next tile's b128 loads issue while
// WMMAs for the current tile run out of LDS.
// ---------------------------------------------------------------------------
__global__ __launch_bounds__(256)
void qkv_gemm(const float* __restrict__ x,
              const float* __restrict__ Wq, const float* __restrict__ bq,
              const float* __restrict__ Wk, const float* __restrict__ bk,
              const float* __restrict__ Wv, const float* __restrict__ bv,
              _Float16* __restrict__ Q, _Float16* __restrict__ K,
              _Float16* __restrict__ V)
{
  const int z = blockIdx.z;
  const float* W    = (z == 0) ? Wq : ((z == 1) ? Wk : Wv);
  const float* bias = (z == 0) ? bq : ((z == 1) ? bk : bv);
  _Float16*    out  = (z == 0) ? Q  : ((z == 1) ? K  : V);

  __shared__ __align__(32) _Float16 sA[128][32];  // x tile  (8 KB)
  __shared__ __align__(32) _Float16 sB[64][32];   // W tile  (4 KB)

  const int tid  = threadIdx.x;
  const int wave = tid >> 5, lane = tid & 31;
  const int row0 = blockIdx.x * 128, col0 = blockIdx.y * 64;
  const int lm = lane & 15, hif = lane >> 4;

  v8f acc[4] = {v8f{}, v8f{}, v8f{}, v8f{}};
  f32x4 ax[4], bx[2];   // staging registers (A: 4 groups, B: 2 groups)

  auto load_tiles = [&](int k0) {
#pragma unroll
    for (int j = 0; j < 4; ++j) {
      int grp = tid + 256 * j;              // 1024 groups of 4 floats
      int r = grp >> 3, kg = (grp & 7) * 4;
      ax[j] = *(const f32x4*)&x[(size_t)(row0 + r) * DM + k0 + kg];
    }
#pragma unroll
    for (int j = 0; j < 2; ++j) {
      int grp = tid + 256 * j;              // 512 groups of 4 floats
      int r = grp >> 3, kg = (grp & 7) * 4;
      bx[j] = *(const f32x4*)&W[(size_t)(col0 + r) * DM + k0 + kg];
    }
  };
  auto store_tiles = [&]() {
#pragma unroll
    for (int j = 0; j < 4; ++j) {
      int grp = tid + 256 * j;
      int r = grp >> 3, kg = (grp & 7) * 4;
      *(v4h*)&sA[r][kg] = cvt4(ax[j]);
    }
#pragma unroll
    for (int j = 0; j < 2; ++j) {
      int grp = tid + 256 * j;
      int r = grp >> 3, kg = (grp & 7) * 4;
      *(v4h*)&sB[r][kg] = cvt4(bx[j]);
    }
  };

  load_tiles(0);
  for (int k0 = 0; k0 < DM; k0 += 32) {
    store_tiles();
    __syncthreads();
    if (k0 + 32 < DM) load_tiles(k0 + 32);   // overlap with WMMAs below

    const int abase = hif ? 16 : 0;    // A frag K half (contiguous, matches B)
    const int kb    = hif ? 16 : 0;    // B frag K half (contiguous, matches A)
    v16h a = *(const v16h*)&sA[wave * 16 + lm][abase];
#pragma unroll
    for (int ct = 0; ct < 4; ++ct) {
      v16h bfr = *(const v16h*)&sB[ct * 16 + lm][kb];
      acc[ct] = wmma16(a, bfr, acc[ct]);
    }
    __syncthreads();
  }

#pragma unroll
  for (int ct = 0; ct < 4; ++ct) {
    int d = col0 + ct * 16 + lm;
    float bvl = bias[d];
    int hsel = d >> 6, c = d & 63;
    if (z == 2) {  // V: [B,H,dk,L]; 8 rows are contiguous along L -> packed
      int g0r = row0 + wave * 16 + hif * 8;
      int bsel = g0r >> 11, l0 = g0r & (SEQ - 1);
      v8h pk;
#pragma unroll
      for (int r = 0; r < 8; ++r) pk[r] = (_Float16)(acc[ct][r] + bvl);
      *(v8h*)&out[(((size_t)(bsel * HEADS + hsel)) * DK + c) * SEQ + l0] = pk;
    } else {       // Q,K: [B,H,L,dk]
#pragma unroll
      for (int r = 0; r < 8; ++r) {
        int g = row0 + wave * 16 + r + hif * 8;
        int bsel = g >> 11, l = g & (SEQ - 1);
        out[(((size_t)(bsel * HEADS + hsel)) * SEQ + l) * DK + c] =
            (_Float16)(acc[ct][r] + bvl);
      }
    }
  }
}

// ---------------------------------------------------------------------------
// Kernel 2: flash attention + closed-form intervention.
// One 16-row q tile per block; keys chunked by 256 with running max/sum/w0.
// V^T chunk (64 x 256, f16) is DMA'd into LDS by the Tensor Data Mover,
// overlapping with the QK^T WMMAs + softmax of the same chunk.
// out = (1-l)*O/S + l*(O - w0*V0)/max(S-w0, 1e-9*S)
// ---------------------------------------------------------------------------
__global__ __launch_bounds__(256)
void attn_kernel(const _Float16* __restrict__ Q, const _Float16* __restrict__ K,
                 const _Float16* __restrict__ Vt, _Float16* __restrict__ A)
{
  const int q0 = blockIdx.x * 16;
  const int h  = blockIdx.y;
  const int b  = blockIdx.z;
  const _Float16* Qb = Q  + ((size_t)(b * HEADS + h) * SEQ) * DK;
  const _Float16* Kb = K  + ((size_t)(b * HEADS + h) * SEQ) * DK;
  const _Float16* Vb = Vt + ((size_t)(b * HEADS + h) * DK) * SEQ; // [dk][SEQ]

  __shared__ __align__(32) float    sS[16][256];    // score chunk   (16 KB)
  __shared__ __align__(32) _Float16 sP[16][256];    // prob chunk    ( 8 KB)
  __shared__ __align__(32) _Float16 sVt[64][256];   // V^T chunk     (32 KB)
  __shared__ float sM[16], sSum[16], sW0[16], sR[16];
  __shared__ float sRed[4][16][16];                 // partial O     ( 4 KB)

  const int tid  = threadIdx.x;
  const int wave = tid >> 5, lane = tid & 31;
  const int lm = lane & 15, hif = lane >> 4;

  if (tid < 16) { sM[tid] = -1e30f; sSum[tid] = 0.f; sW0[tid] = 0.f; }

  // Q A-fragments (constant over key loop), pre-scaled by 1/sqrt(dk)=0.125
  const int abase = hif ? 16 : 0;
  v16h aq[2];
#pragma unroll
  for (int ks = 0; ks < 2; ++ks) {
    const _Float16* qp = Qb + (size_t)(q0 + lm) * DK + ks * 32;
    v16h t = *(const v16h*)(qp + abase);
#pragma unroll
    for (int i = 0; i < 16; ++i) t[i] *= (_Float16)0.125f;
    aq[ks] = t;
  }

  v8f acc = {};              // O accumulator, 16x16 fragment per wave
  const int nt = wave & 3;   // dk column tile (0..3)
  const int kh = wave >> 2;  // key half within chunk (0..1)

  __syncthreads();

  for (int kc = 0; kc < SEQ / 256; ++kc) {
    const int kb0 = kc * 256;
    const int kb  = hif ? 16 : 0;

    // phase 1: start async staging of V^T chunk into LDS
#if USE_TDM
    if (wave == 0)
      tdm_load_tile(Vb + kb0, (unsigned)(uintptr_t)&sVt[0][0]);
#else
#pragma unroll
    for (int j = 0; j < 8; ++j) {
      int id = tid + 256 * j;             // 2048 groups of 8 halfs
      int rrow = id >> 5, grp = (id & 31) * 8;
      *(v8h*)&sVt[rrow][grp] =
          *(const v8h*)&Vb[(size_t)rrow * SEQ + kb0 + grp];
    }
#endif

    // phase 2: scores = Q * K^T for this wave's two 16-key tiles
#pragma unroll
    for (int ct2 = 0; ct2 < 2; ++ct2) {
      int ct = wave * 2 + ct2;
      v8f sc = {};
#pragma unroll
      for (int ks = 0; ks < 2; ++ks) {
        const _Float16* kp =
            Kb + (size_t)(kb0 + ct * 16 + lm) * DK + ks * 32 + kb;
        v16h bk = *(const v16h*)kp;
        __builtin_prefetch(kp + 256 * DK, 0, 1);   // next key chunk
        sc = wmma16(aq[ks], bk, sc);
      }
#pragma unroll
      for (int r = 0; r < 8; ++r) sS[r + hif * 8][ct * 16 + lm] = sc[r];
    }
    __syncthreads();

    // phase 3: chunk softmax with running max (16 threads per q row)
    {
      int row = tid >> 4, sub = tid & 15;
      float mx = -1e30f;
#pragma unroll
      for (int j = 0; j < 16; ++j) mx = fmaxf(mx, sS[row][sub + 16 * j]);
#pragma unroll
      for (int off = 8; off >= 1; off >>= 1)
        mx = fmaxf(mx, __shfl_xor(mx, off, 16));
      float mold = sM[row];
      float mnew = fmaxf(mold, mx);
      float p0 = 0.f, ssum = 0.f;
#pragma unroll
      for (int j = 0; j < 16; ++j) {
        int c = sub + 16 * j;
        float p = __expf(sS[row][c] - mnew);
        sP[row][c] = (_Float16)p;
        ssum += p;
        if (kc == 0 && c == 0) p0 = p;   // unnormalized weight of key 0
      }
#pragma unroll
      for (int off = 8; off >= 1; off >>= 1) ssum += __shfl_xor(ssum, off, 16);
      if (sub == 0) {
        float rs = __expf(mold - mnew);
        sR[row]   = rs;
        sM[row]   = mnew;
        sSum[row] = sSum[row] * rs + ssum;
        sW0[row]  = sW0[row] * rs + p0;
      }
    }
#if USE_TDM
    if (wave == 0) __builtin_amdgcn_s_wait_tensorcnt(0);  // V^T chunk landed
#endif
    __syncthreads();

    // phase 4: rescale accumulator, then O += P * V over this wave's key half
#pragma unroll
    for (int r = 0; r < 8; ++r) acc[r] *= sR[r + hif * 8];
#pragma unroll
    for (int ks = 0; ks < 4; ++ks) {
      int ck = kh * 128 + ks * 32;
      v16h ap = *(const v16h*)&sP[lm][ck + abase];
      v16h bv = *(const v16h*)&sVt[nt * 16 + lm][ck + kb];
      acc = wmma16(ap, bv, acc);
    }
    __syncthreads();
  }

  // epilogue: combine key halves, apply intervention, store f16 attn output
  if (kh == 1) {
#pragma unroll
    for (int r = 0; r < 8; ++r) sRed[nt][r + hif * 8][lm] = acc[r];
  }
  __syncthreads();
  if (kh == 0) {
    int dcol = nt * 16 + lm;
    float v0 = (float)Vb[(size_t)dcol * SEQ];   // V at key 0
#pragma unroll
    for (int r = 0; r < 8; ++r) {
      int m = r + hif * 8;
      float On = acc[r] + sRed[nt][m][lm];
      float S  = sSum[m];
      float W0 = sW0[m];
      float o1 = On / S;
      float o2 = (On - W0 * v0) / fmaxf(S - W0, 1e-9f * S);
      float o  = (1.f - LAM) * o1 + LAM * o2;
      size_t tok = (size_t)b * SEQ + q0 + m;
      A[tok * DM + h * DK + dcol] = (_Float16)o;
    }
  }
}

// ---------------------------------------------------------------------------
// Kernel 3: output projection. out = attn @ Wo^T + bo (f32 output)
// ---------------------------------------------------------------------------
__global__ __launch_bounds__(256)
void oproj_gemm(const _Float16* __restrict__ Ain, const float* __restrict__ Wo,
                const float* __restrict__ bo, float* __restrict__ out)
{
  __shared__ __align__(32) _Float16 sA[128][32];
  __shared__ __align__(32) _Float16 sB[64][32];
  const int tid  = threadIdx.x;
  const int wave = tid >> 5, lane = tid & 31;
  const int row0 = blockIdx.x * 128, col0 = blockIdx.y * 64;
  const int lm = lane & 15, hif = lane >> 4;

  v8f acc[4] = {v8f{}, v8f{}, v8f{}, v8f{}};
  v8h   aa[2];   // A staging (f16, b128 moves)
  f32x4 bx[2];   // B staging (f32 -> f16)

  auto load_tiles = [&](int k0) {
#pragma unroll
    for (int j = 0; j < 2; ++j) {
      int grp = tid + 256 * j;              // 512 groups of 8 halfs
      int r = grp >> 2, kg = (grp & 3) * 8;
      aa[j] = *(const v8h*)&Ain[(size_t)(row0 + r) * DM + k0 + kg];
    }
#pragma unroll
    for (int j = 0; j < 2; ++j) {
      int grp = tid + 256 * j;              // 512 groups of 4 floats
      int r = grp >> 3, kg = (grp & 7) * 4;
      bx[j] = *(const f32x4*)&Wo[(size_t)(col0 + r) * DM + k0 + kg];
    }
  };
  auto store_tiles = [&]() {
#pragma unroll
    for (int j = 0; j < 2; ++j) {
      int grp = tid + 256 * j;
      int r = grp >> 2, kg = (grp & 3) * 8;
      *(v8h*)&sA[r][kg] = aa[j];
    }
#pragma unroll
    for (int j = 0; j < 2; ++j) {
      int grp = tid + 256 * j;
      int r = grp >> 3, kg = (grp & 7) * 4;
      *(v4h*)&sB[r][kg] = cvt4(bx[j]);
    }
  };

  load_tiles(0);
  for (int k0 = 0; k0 < DM; k0 += 32) {
    store_tiles();
    __syncthreads();
    if (k0 + 32 < DM) load_tiles(k0 + 32);

    const int abase = hif ? 16 : 0, kb = hif ? 16 : 0;
    v16h a = *(const v16h*)&sA[wave * 16 + lm][abase];
#pragma unroll
    for (int ct = 0; ct < 4; ++ct) {
      v16h bfr = *(const v16h*)&sB[ct * 16 + lm][kb];
      acc[ct] = wmma16(a, bfr, acc[ct]);
    }
    __syncthreads();
  }
#pragma unroll
  for (int ct = 0; ct < 4; ++ct) {
    int d = col0 + ct * 16 + lm;
    float bvl = bo[d];
#pragma unroll
    for (int r = 0; r < 8; ++r) {
      int g = row0 + wave * 16 + r + hif * 8;
      out[(size_t)g * DM + d] = acc[ct][r] + bvl;
    }
  }
}

extern "C" void kernel_launch(void* const* d_in, const int* in_sizes, int n_in,
                              void* d_out, int out_size, void* d_ws, size_t ws_size,
                              hipStream_t stream)
{
  (void)in_sizes; (void)n_in; (void)out_size; (void)ws_size;
  const float* x  = (const float*)d_in[0];
  const float* Wq = (const float*)d_in[1];
  const float* bq = (const float*)d_in[2];
  const float* Wk = (const float*)d_in[3];
  const float* bk = (const float*)d_in[4];
  const float* Wv = (const float*)d_in[5];
  const float* bv = (const float*)d_in[6];
  const float* Wo = (const float*)d_in[7];
  const float* bo = (const float*)d_in[8];

  // workspace: Q,K [B,H,L,dk], V^T [B,H,dk,L], attn [N,D]; all f16 = 32 MB
  _Float16* Qws = (_Float16*)d_ws;
  _Float16* Kws = Qws + (size_t)NTOK * DM;
  _Float16* Vws = Kws + (size_t)NTOK * DM;
  _Float16* Aws = Vws + (size_t)NTOK * DM;

  qkv_gemm<<<dim3(NTOK / 128, DM / 64, 3), 256, 0, stream>>>(
      x, Wq, bq, Wk, bk, Wv, bv, Qws, Kws, Vws);
  attn_kernel<<<dim3(SEQ / 16, HEADS, BATCH), 256, 0, stream>>>(
      Qws, Kws, Vws, Aws);
  oproj_gemm<<<dim3(NTOK / 128, DM / 64), 256, 0, stream>>>(
      Aws, Wo, bo, (float*)d_out);
}
